// LinearAttention_50989851738187
// MI455X (gfx1250) — matrix operand order repaired and checked by reading, not compiled, not run on hardware
//
#include <hip/hip_runtime.h>
#include <hip/hip_bf16.h>

#define DEVI __device__ __forceinline__

typedef _Float16 v16h __attribute__((ext_vector_type(16)));
typedef _Float16 v8h  __attribute__((ext_vector_type(8)));
typedef float    v8f  __attribute__((ext_vector_type(8)));

constexpr int   B_    = 2;
constexpr int   C_    = 128;
constexpr int   HEADS = 8;
constexpr int   CPG   = 16;
constexpr int   NTOK  = 16 * 96 * 96;      // 147456 tokens per batch
constexpr int   NTILE = NTOK / 16;         // 9216 16-token tiles per batch
constexpr float EPSLN = 1e-5f;

constexpr int   PAD16 = 24;   // padded row (halves) for 16-half LDS tiles (48B, 16B-aligned)
constexpr int   PADC  = 136;  // padded row (halves) for 128-half LDS tiles (272B, 16B-aligned)
constexpr int   PADQ  = 20;   // padded row (floats) for q LDS tile (80B, 16B-aligned)

struct WS {
  _Float16* wk_h;   // [8][16(o)][16(k)] f16, g_pre folded
  _Float16* wv_h;
  _Float16* wq_h;
  _Float16* wo_h;   // [128][128] f16
  _Float16* ctxT_h; // [B][8][16(e)][16(d)] f16, scale/Z folded
  float* Sk; float* Sv; float* Sq;   // [8][16] row sums of rounded weights
  float* Z;         // [B][8][16]  (indexed by d)
  float* ctx_un;    // [B][8][16(e)][16(d)]
  float* meanb;     // [B][NTOK]
  float* rstdb;     // [B][NTOK]
  unsigned* kmax_e; // [B][8][16] order-encoded float
};

DEVI int rfl(int v) { return __builtin_amdgcn_readfirstlane(v); }

DEVI v8f zero8() { v8f z;
#pragma unroll
  for (int i = 0; i < 8; ++i) z[i] = 0.f;
  return z; }

DEVI v8f wmma16(v16h a, v16h b, v8f c) {
  return __builtin_amdgcn_wmma_f32_16x16x32_f16(false, a, false, b, (short)0, c, false, false);
}

// ---- fragment builders (CDNA5 wave32 layouts) -------------------------------
// A 16x32 f16: lane m=l&15, kh=l>>4; elems 0..7 -> K=kh*8+e, elems 8..15 -> K=16+kh*8+e
// B 32x16 f16: lane n=l&15, kh=l>>4; elems 0..7 -> K=kh*8+e, elems 8..15 -> K=16+kh*8+e
// C/D 16x16 f32: reg r -> row r+8*kh, col = l&15

// A fragment from a 16-half row; real K = 0..15, upper K zero-padded.
DEVI v16h afrag_row16(const _Float16* row, int kh) {
  v8h lo = *(const v8h*)(row + kh * 8);
  v16h a;
#pragma unroll
  for (int i = 0; i < 8; ++i) { a[i] = lo[i]; a[8 + i] = (_Float16)0.f; }
  return a;
}

// B fragment from a 16-half row (one column of B, K-major); K 16..31 are zero.
DEVI v16h bfrag_row16(const _Float16* row, int kh) {
  v8h lo = *(const v8h*)(row + kh * 8);
  v16h b;
#pragma unroll
  for (int i = 0; i < 8; ++i) { b[i] = lo[i]; b[8 + i] = (_Float16)0.f; }
  return b;
}

// A fragment, full K=32, from a row-major 32-half row chunk.
DEVI v16h afrag_k32(const _Float16* row, int kh) {
  v8h lo = *(const v8h*)(row + kh * 8);
  v8h hi = *(const v8h*)(row + 16 + kh * 8);
  v16h a;
#pragma unroll
  for (int i = 0; i < 8; ++i) { a[i] = lo[i]; a[8 + i] = hi[i]; }
  return a;
}

// B fragment, full K=32, from a 32-half contiguous run (one column of B).
DEVI v16h bfrag_k32(const _Float16* row, int kh) {
  v8h lo = *(const v8h*)(row + kh * 8);
  v8h hi = *(const v8h*)(row + 16 + kh * 8);
  v16h b;
#pragma unroll
  for (int i = 0; i < 8; ++i) { b[i] = lo[i]; b[8 + i] = hi[i]; }
  return b;
}

// A fragment of x^T: lane = token row, element e -> channel kh*8+e (K>=16 zero).
// p points at x[ch0*NTOK + token]; no cross-lane traffic, full-wave loads.
DEVI v16h afrag_xT(const float* p, int kh) {
  v16h a;
#pragma unroll
  for (int e = 0; e < 8; ++e) {
    a[e]     = (_Float16)p[(size_t)(kh * 8 + e) * NTOK];
    a[8 + e] = (_Float16)0.f;
  }
  return a;
}

// ---- order-preserving float <-> uint encoding for atomicMax -----------------
DEVI unsigned encf(float f) {
  unsigned u = __float_as_uint(f);
  return u ^ ((unsigned)((int)u >> 31) | 0x80000000u);
}
DEVI float decf(unsigned e) {
  unsigned u = (e & 0x80000000u) ? (e ^ 0x80000000u) : ~e;
  return __uint_as_float(u);
}

// =============================================================================
// K0: weight prep. threads 0..383: fold g_pre into wk/wv/wq rows, f16 round,
// row sums of the rounded weights. 384..511: wo -> f16. 512..1023: zero bufs.
__global__ __launch_bounds__(256) void k0_prep(const float* wq, const float* wk,
                                               const float* wv, const float* wo,
                                               const float* gpre, WS ws) {
  int id = blockIdx.x * blockDim.x + threadIdx.x;
  if (id < 384) {
    int which = id >> 7, row = id & 127;
    const float* src = (which == 0) ? wk : (which == 1) ? wv : wq;
    _Float16* dst = (which == 0) ? ws.wk_h : (which == 1) ? ws.wv_h : ws.wq_h;
    float* S = (which == 0) ? ws.Sk : (which == 1) ? ws.Sv : ws.Sq;
    int h = row >> 4;
    float s = 0.f;
#pragma unroll
    for (int i = 0; i < 16; ++i) {
      _Float16 hv = (_Float16)(src[row * 16 + i] * gpre[h * 16 + i]);
      dst[row * 16 + i] = hv;
      s += (float)hv;
    }
    S[row] = s;
  } else if (id < 512) {
    int row = id - 384;
    for (int c = 0; c < 128; ++c) ws.wo_h[row * 128 + c] = (_Float16)wo[row * 128 + c];
  } else {
    int zid = id - 512;                       // 512 threads
    for (int k = zid; k < B_ * HEADS * 256; k += 512) ws.ctx_un[k] = 0.f;
    if (zid < B_ * HEADS * 16) { ws.kmax_e[zid] = 0u; ws.Z[zid] = 0.f; }
  }
}

// K1: per-token LayerNorm stats (mean, rstd) over 128 channels.
__global__ __launch_bounds__(256) void k1_stats(const float* x, WS ws) {
  int id = blockIdx.x * blockDim.x + threadIdx.x;
  if (id >= B_ * NTOK) return;
  int bi = id / NTOK, t = id - bi * NTOK;
  const float* p = x + (size_t)bi * C_ * NTOK + t;
  float s = 0.f, q = 0.f;
#pragma unroll 4
  for (int c = 0; c < C_; ++c) { float v = p[(size_t)c * NTOK]; s += v; q += v * v; }
  float m = s * (1.f / C_);
  float var = q * (1.f / C_) - m * m;
  ws.meanb[id] = m;
  ws.rstdb[id] = rsqrtf(var + EPSLN);
}

// K2: per-(b,h,d) max of k_raw^T[t][o] = rstd_t*( (x^T W^T)[t][o] - mean_t*S[o] ).
__global__ __launch_bounds__(256) void k2_kmax(const float* x, WS ws) {
  const int bi = blockIdx.y;
  const int lane = threadIdx.x & 31;
  const int wv = rfl(threadIdx.x >> 5);
  const int kh = lane >> 4, ln = lane & 15;
  const int gw = rfl(blockIdx.x * 8 + wv), nw = rfl(gridDim.x * 8);
  const float* xb = x + (size_t)bi * C_ * NTOK;
  const float* mb = ws.meanb + bi * NTOK;
  const float* rb = ws.rstdb + bi * NTOK;
  for (int h = 0; h < HEADS; ++h) {
    v16h bw = bfrag_row16(ws.wk_h + (h * 16 + ln) * 16, kh); // B = W'^T, col o=ln
    float Sl = ws.Sk[h * 16 + ln];
    float kmx = -3.0e38f;
    for (int t = gw; t < NTILE; t += nw) {
      int t16 = t * 16;
      float mrow[8], rrow[8];
      const float* mp = mb + t16 + kh * 8;
      const float* rp = rb + t16 + kh * 8;
#pragma unroll
      for (int r = 0; r < 8; ++r) { mrow[r] = mp[r]; rrow[r] = rp[r]; }
      v16h ax = afrag_xT(xb + (size_t)(h * 16) * NTOK + t16 + ln, kh);
      v8f d = wmma16(ax, bw, zero8());     // D[t=r+8kh][o=ln]
#pragma unroll
      for (int r = 0; r < 8; ++r) kmx = fmaxf(kmx, rrow[r] * (d[r] - mrow[r] * Sl));
    }
    kmx = fmaxf(kmx, __shfl_xor(kmx, 16));
    if (kh == 0) atomicMax(&ws.kmax_e[(bi * HEADS + h) * 16 + ln], encf(kmx));
  }
}

// K3: Z[d] = sum_t exp(k-kmax); ctxT[e][d] = sum_t v[e,t]*p[t,d] (WMMA, K=tokens).
__global__ __launch_bounds__(256) void k3_ctx(const float* x, WS ws) {
  __shared__ __align__(16) _Float16 vsA[8][16][PAD16]; // [wave][e][t]  (A = V^T)
  __shared__ __align__(16) _Float16 psB[8][16][PAD16]; // [wave][d][t]  (B = P[t][d] cols)
  __shared__ float comb[8][256];
  __shared__ float zcomb[8][16];
  const int bi = blockIdx.y;
  const int lane = threadIdx.x & 31;
  const int wv = rfl(threadIdx.x >> 5);
  const int kh = lane >> 4, ln = lane & 15;
  const int gw = rfl(blockIdx.x * 8 + wv), nw = rfl(gridDim.x * 8);
  const float* xb = x + (size_t)bi * C_ * NTOK;
  const float* mb = ws.meanb + bi * NTOK;
  const float* rb = ws.rstdb + bi * NTOK;
  for (int h = 0; h < HEADS; ++h) {
    v16h bk = bfrag_row16(ws.wk_h + (h * 16 + ln) * 16, kh);
    v16h bv = bfrag_row16(ws.wv_h + (h * 16 + ln) * 16, kh);
    float Skl = ws.Sk[h * 16 + ln];
    float Svl = ws.Sv[h * 16 + ln];
    float kml = decf(ws.kmax_e[(bi * HEADS + h) * 16 + ln]);
    float zacc = 0.f;
    v8f cacc = zero8();
    for (int t = gw; t < NTILE; t += nw) {
      int t16 = t * 16;
      float mrow[8], rrow[8];
      const float* mp = mb + t16 + kh * 8;
      const float* rp = rb + t16 + kh * 8;
#pragma unroll
      for (int r = 0; r < 8; ++r) { mrow[r] = mp[r]; rrow[r] = rp[r]; }
      v16h ax = afrag_xT(xb + (size_t)(h * 16) * NTOK + t16 + ln, kh);
      v8f dk = wmma16(ax, bk, zero8());    // k^T[t][d=ln]
      v8f dv = wmma16(ax, bv, zero8());    // v^T[t][e=ln]
      v8h pv, vvv;
#pragma unroll
      for (int r = 0; r < 8; ++r) {
        float p = __expf(rrow[r] * (dk[r] - mrow[r] * Skl) - kml);
        zacc += p;
        pv[r]  = (_Float16)p;
        vvv[r] = (_Float16)(rrow[r] * (dv[r] - mrow[r] * Svl));
      }
      *(v8h*)(&psB[wv][ln][kh * 8]) = pv;   // P[t=kh*8+r][d=ln] -> row d=ln
      *(v8h*)(&vsA[wv][ln][kh * 8]) = vvv;  // V^T[e=ln][t=kh*8+r]
      __builtin_amdgcn_wave_barrier();      // DS write -> read ordering
      v16h aV = afrag_row16(&vsA[wv][ln][0], kh);
      v16h bP = bfrag_row16(&psB[wv][ln][0], kh);
      cacc = wmma16(aV, bP, cacc);          // ctxT[e=r+8kh][d=ln]
      __builtin_amdgcn_wave_barrier();
    }
    zacc += __shfl_xor(zacc, 16);
    if (kh == 0) zcomb[wv][ln] = zacc;
#pragma unroll
    for (int r = 0; r < 8; ++r) comb[wv][(r + 8 * kh) * 16 + ln] = cacc[r];
    __syncthreads();
    {
      int tid = threadIdx.x;
      float s = 0.f;
#pragma unroll
      for (int w = 0; w < 8; ++w) s += comb[w][tid];
      atomicAdd(&ws.ctx_un[(bi * HEADS + h) * 256 + tid], s);  // [e][d]
      if (tid < 16) {
        float z = 0.f;
#pragma unroll
        for (int w = 0; w < 8; ++w) z += zcomb[w][tid];
        atomicAdd(&ws.Z[(bi * HEADS + h) * 16 + tid], z);      // Z[d]
      }
    }
    __syncthreads();
  }
}

// K4: ctxT_h[e][d] = 0.25 * ctx_un[e][d] / Z[d]   (q-softmax scale folded)
__global__ __launch_bounds__(256) void k4_fin(WS ws) {
  int id = blockIdx.x * blockDim.x + threadIdx.x;
  if (id >= B_ * HEADS * 256) return;
  int d = id & 15, e = (id >> 4) & 15, hh = (id >> 8) & 7, bi = id >> 11;
  float z = ws.Z[(bi * HEADS + hh) * 16 + d];
  float c = ws.ctx_un[(bi * HEADS + hh) * 256 + e * 16 + d];
  ws.ctxT_h[((bi * HEADS + hh) * 16 + e) * 16 + d] = (_Float16)(0.25f * c / z);
}

// K5: q -> softmax(d) -> out = ctxT*q -> y = wo*out + bo -> LN -> +x -> store.
__global__ __launch_bounds__(128) void k5_out(const float* x, const float* bo,
                                              const float* gout, float* y, WS ws) {
  __shared__ __align__(16) _Float16 wos[128][PADC];    // staged f16 wo, padded rows
  __shared__ __align__(16) _Float16 outs[4][16][PADC]; // [wave][t][c]
  __shared__ __align__(16) float    qbuf[4][16][PADQ]; // [wave][t][d] raw q (f32)
  const int bi = blockIdx.y;
  const int lane = threadIdx.x & 31;
  const int wv = rfl(threadIdx.x >> 5);
  const int kh = lane >> 4, ln = lane & 15;
  const int gw = rfl(blockIdx.x * 4 + wv), nw = rfl(gridDim.x * 4);
  const float* xb = x + (size_t)bi * C_ * NTOK;
  const float* mb = ws.meanb + bi * NTOK;
  const float* rb = ws.rstdb + bi * NTOK;
  float* yb = y + (size_t)bi * C_ * NTOK;

  // stage wo (f16, 32KB) into LDS once per block; one row per thread
  {
    int row = threadIdx.x;  // 128 threads, 128 rows
    const v8h* src = (const v8h*)(ws.wo_h + row * 128);
    v8h* dst = (v8h*)(&wos[row][0]);
#pragma unroll
    for (int j = 0; j < 16; ++j) dst[j] = src[j];
  }
  __syncthreads();

  for (int t = gw; t < NTILE; t += nw) {
    int t16 = t * 16;
    int tok = t16 + ln;
    float mrow[8], rrow[8];
    {
      const float* mp = mb + t16 + kh * 8;
      const float* rp = rb + t16 + kh * 8;
#pragma unroll
      for (int r = 0; r < 8; ++r) { mrow[r] = mp[r]; rrow[r] = rp[r]; }
    }
    // ---- per-head: q^T via WMMA, per-token serial softmax, out rows ----
    for (int h = 0; h < HEADS; ++h) {
      v16h bq = bfrag_row16(ws.wq_h + (h * 16 + ln) * 16, kh);
      float Sql = ws.Sq[h * 16 + ln];
      v16h ax = afrag_xT(xb + (size_t)(h * 16) * NTOK + t16 + ln, kh);
      v8f dq = wmma16(ax, bq, zero8());    // q^T[t=r+8kh][d=ln]
#pragma unroll
      for (int r = 0; r < 8; ++r)
        qbuf[wv][r + 8 * kh][ln] = rrow[r] * (dq[r] - mrow[r] * Sql);
      __builtin_amdgcn_wave_barrier();
      // each lane reads the full 16-d row of token ln and does a serial softmax
      float qv[16];
      {
        const float4* q4 = (const float4*)(&qbuf[wv][ln][0]);
#pragma unroll
        for (int j = 0; j < 4; ++j) {
          float4 v = q4[j];
          qv[4 * j] = v.x; qv[4 * j + 1] = v.y; qv[4 * j + 2] = v.z; qv[4 * j + 3] = v.w;
        }
      }
      float mx = qv[0];
#pragma unroll
      for (int i = 1; i < 16; ++i) mx = fmaxf(mx, qv[i]);
      float ssum = 0.f;
#pragma unroll
      for (int i = 0; i < 16; ++i) { qv[i] = __expf(qv[i] - mx); ssum += qv[i]; }
      float inv = 1.f / ssum;
      v16h bqn;
#pragma unroll
      for (int e = 0; e < 16; ++e)
        bqn[e] = (e < 8) ? (_Float16)(qv[kh * 8 + e] * inv) : (_Float16)0.f;
      __builtin_amdgcn_wave_barrier();     // reads done before next head's writes
      v16h actx = afrag_row16(ws.ctxT_h + ((bi * HEADS + h) * 16 + ln) * 16, kh);
      v8f of = wmma16(actx, bqn, zero8()); // out[e=r+8kh][t=ln]
      v8h oh;
#pragma unroll
      for (int r = 0; r < 8; ++r) oh[r] = (_Float16)of[r];
      *(v8h*)(&outs[wv][ln][h * 16 + kh * 8]) = oh;  // [t][c], packed b128
    }
    __builtin_amdgcn_wave_barrier();
    // ---- wo matmul: y[o][t] = sum_c wo[o][c]*outs[t][c], K=128 (4 WMMA chunks) ----
    v8f acc[8];
#pragma unroll
    for (int ot = 0; ot < 8; ++ot) acc[ot] = zero8();
#pragma unroll
    for (int c0 = 0; c0 < 128; c0 += 32) {
      v16h bfo = bfrag_k32(&outs[wv][ln][c0], kh);
#pragma unroll
      for (int ot = 0; ot < 8; ++ot) {
        v16h aw = afrag_k32(&wos[ot * 16 + ln][c0], kh);
        acc[ot] = wmma16(aw, bfo, acc[ot]);
      }
    }
    // ---- bias + output LayerNorm stats ----
    float sum = 0.f, sq2 = 0.f;
#pragma unroll
    for (int ot = 0; ot < 8; ++ot) {
      const float* bp = bo + ot * 16 + kh * 8;
#pragma unroll
      for (int r = 0; r < 8; ++r) {
        float yv = acc[ot][r] + bp[r];
        acc[ot][r] = yv;
        sum += yv;
        sq2 += yv * yv;
      }
    }
    sum += __shfl_xor(sum, 16);
    sq2 += __shfl_xor(sq2, 16);
    float mean2 = sum * (1.f / C_);
    float rstd2 = rsqrtf(sq2 * (1.f / C_) - mean2 * mean2 + EPSLN);
    // ---- scale + residual + store ----
#pragma unroll
    for (int ot = 0; ot < 8; ++ot) {
      const float* gp = gout + ot * 16 + kh * 8;
      const float* xr = xb + (size_t)(ot * 16 + kh * 8) * NTOK + tok;
      float* yr = yb + (size_t)(ot * 16 + kh * 8) * NTOK + tok;
#pragma unroll
      for (int r = 0; r < 8; ++r)
        yr[(size_t)r * NTOK] = (acc[ot][r] - mean2) * rstd2 * gp[r] + xr[(size_t)r * NTOK];
    }
  }
}

// =============================================================================
extern "C" void kernel_launch(void* const* d_in, const int* in_sizes, int n_in,
                              void* d_out, int out_size, void* d_ws, size_t ws_size,
                              hipStream_t stream) {
  const float* x    = (const float*)d_in[0];
  const float* wq   = (const float*)d_in[1];
  const float* wk   = (const float*)d_in[2];
  const float* wv   = (const float*)d_in[3];
  const float* wo   = (const float*)d_in[4];
  const float* bo   = (const float*)d_in[5];
  const float* gpre = (const float*)d_in[6];
  const float* gout = (const float*)d_in[7];
  float* out = (float*)d_out;

  char* base = (char*)d_ws;
  size_t off = 0;
  auto take = [&](size_t bytes) -> void* {
    void* p = base + off;
    off = (off + bytes + 255) & ~(size_t)255;
    return p;
  };
  WS ws;
  ws.wk_h   = (_Float16*)take(HEADS * CPG * CPG * sizeof(_Float16));
  ws.wv_h   = (_Float16*)take(HEADS * CPG * CPG * sizeof(_Float16));
  ws.wq_h   = (_Float16*)take(HEADS * CPG * CPG * sizeof(_Float16));
  ws.wo_h   = (_Float16*)take((size_t)C_ * C_ * sizeof(_Float16));
  ws.ctxT_h = (_Float16*)take((size_t)B_ * HEADS * 256 * sizeof(_Float16));
  ws.Sk     = (float*)take(HEADS * CPG * sizeof(float));
  ws.Sv     = (float*)take(HEADS * CPG * sizeof(float));
  ws.Sq     = (float*)take(HEADS * CPG * sizeof(float));
  ws.Z      = (float*)take((size_t)B_ * HEADS * 16 * sizeof(float));
  ws.ctx_un = (float*)take((size_t)B_ * HEADS * 256 * sizeof(float));
  ws.kmax_e = (unsigned*)take((size_t)B_ * HEADS * 16 * sizeof(unsigned));
  ws.meanb  = (float*)take((size_t)B_ * NTOK * sizeof(float));
  ws.rstdb  = (float*)take((size_t)B_ * NTOK * sizeof(float));
  (void)in_sizes; (void)n_in; (void)out_size; (void)ws_size;

  k0_prep<<<4, 256, 0, stream>>>(wq, wk, wv, wo, gpre, ws);
  k1_stats<<<(B_ * NTOK + 255) / 256, 256, 0, stream>>>(x, ws);
  k2_kmax<<<dim3(64, B_), 256, 0, stream>>>(x, ws);
  k3_ctx<<<dim3(64, B_), 256, 0, stream>>>(x, ws);
  k4_fin<<<(B_ * HEADS * 256 + 255) / 256, 256, 0, stream>>>(ws);
  k5_out<<<dim3(96, B_), 128, 0, stream>>>(x, bo, gout, out, ws);
}
